// MultiHeadAttention_26414048870612
// MI455X (gfx1250) — matrix operand order repaired and checked by reading, not compiled, not run on hardware
//
#include <hip/hip_runtime.h>
#include <hip/hip_bf16.h>

// ---------------------------------------------------------------------------
// MHA for MI455X (gfx1250, wave32, WMMA, 320KB LDS).
//   B=1, S=2048, D=1024, H=16, DK=64
// Pipeline:
//   1) f32->f16 convert (activations + weights)
//   2) WMMA projection GEMMs (software-pipelined K loop, 16x64 per wave)
//      Q,K -> [h][s][dk] ; V -> [h][dk][s]
//   3) fused attention per (head, 16-row tile):
//        scores (WMMA) -> 128KB LDS tile -> softmax in LDS (shfl reductions)
//        -> attn weights written once to d_out -> AV (WMMA, A from LDS,
//        cross-wave ds_add_f32 accumulation) -> ctx f16
//   4) WMMA output projection, f32 + bias -> d_out
// ---------------------------------------------------------------------------

typedef __attribute__((ext_vector_type(16))) _Float16 v16h;
typedef __attribute__((ext_vector_type(8)))  float    v8f;

#define S_  2048
#define D_  1024
#define H_  16
#define DK_ 64
#define TPAD 2052   // LDS row stride (floats): 2052 % 64 == 4 -> conflict-free

// ---- fragment loaders (v_wmma_f32_16x16x32_f16 layouts, wave32) -----------

// A: 16x32 f16, row-major source, leading dim lda (halfs).
// lane 0-15: M=lane, K=0..7 / 16..23 ; lane16-31: M=lane-16, K=8..15 / 24..31
__device__ inline v16h load_a_f16(const _Float16* __restrict__ A, int m0,
                                  int k0, int lda, int lane) {
    const int row  = lane & 15;
    const int half = lane >> 4;
    const _Float16* p = A + (size_t)(m0 + row) * lda + k0 + half * 8;
    v16h a;
#pragma unroll
    for (int j = 0; j < 8; ++j) { a[j] = p[j]; a[j + 8] = p[j + 16]; }
    return a;
}

// B: 32x16 f16, storage transposed: Bt[n][k], leading dim ldb (halfs).
// lane 0-15: N=lane, K=0..7 / 16..23 ; lane16-31: N=lane-16, K=8..15 / 24..31
__device__ inline v16h load_bt_f16(const _Float16* __restrict__ Bt, int n0,
                                   int k0, int ldb, int lane) {
    const int col  = lane & 15;
    const int half = lane >> 4;
    const _Float16* p = Bt + (size_t)(n0 + col) * ldb + k0 + half * 8;
    v16h b;
#pragma unroll
    for (int j = 0; j < 8; ++j) { b[j] = p[j]; b[j + 8] = p[j + 16]; }
    return b;
}

// ---- f32 -> f16 conversion -------------------------------------------------

__global__ void cvt_f32_f16(const float* __restrict__ src,
                            _Float16* __restrict__ dst, int n) {
    int i = blockIdx.x * blockDim.x + threadIdx.x;
    int stride = gridDim.x * blockDim.x;
    for (; i < n; i += stride) dst[i] = (_Float16)src[i];
}

// ---- generic WMMA GEMM: C[m,n] = sum_k A16[m,k]*Bt16[n,k] + bias[n] --------
// Wave computes 16x64 (4 accumulators reuse A). K-loop software-pipelined.
// mode 0: f16 out [h][s][dk] ; mode 1: f16 out [h][dk][s] ; mode 2: f32 [m][n]
__global__ void gemm_wmma(const _Float16* __restrict__ A,
                          const _Float16* __restrict__ Bt,
                          const float* __restrict__ bias,
                          _Float16* __restrict__ out16,
                          float* __restrict__ out32,
                          int M, int N, int K, int mode) {
    const int wave = (blockIdx.x * blockDim.x + threadIdx.x) >> 5;
    const int lane = threadIdx.x & 31;
    const int nStrips = N / 64;
    if (wave >= (M / 16) * nStrips) return;  // wave-uniform
    const int m0 = (wave / nStrips) * 16;
    const int n0 = (wave % nStrips) * 64;

    v8f acc[4] = {};
    // prologue: fragments for k0 = 0
    v16h a  = load_a_f16(A, m0, 0, K, lane);
    v16h b0 = load_bt_f16(Bt, n0 +  0, 0, K, lane);
    v16h b1 = load_bt_f16(Bt, n0 + 16, 0, K, lane);
    v16h b2 = load_bt_f16(Bt, n0 + 32, 0, K, lane);
    v16h b3 = load_bt_f16(Bt, n0 + 48, 0, K, lane);

    for (int k0 = 0; k0 < K; k0 += 32) {
        // issue next-tile loads before consuming current fragments
        const int kn = (k0 + 32 < K) ? (k0 + 32) : 0;  // dummy reload last iter
        v16h an  = load_a_f16(A, m0, kn, K, lane);
        v16h c0  = load_bt_f16(Bt, n0 +  0, kn, K, lane);
        v16h c1  = load_bt_f16(Bt, n0 + 16, kn, K, lane);
        v16h c2  = load_bt_f16(Bt, n0 + 32, kn, K, lane);
        v16h c3  = load_bt_f16(Bt, n0 + 48, kn, K, lane);

        acc[0] = __builtin_amdgcn_wmma_f32_16x16x32_f16(false, a, false, b0,
                                                        (short)0, acc[0], false, false);
        acc[1] = __builtin_amdgcn_wmma_f32_16x16x32_f16(false, a, false, b1,
                                                        (short)0, acc[1], false, false);
        acc[2] = __builtin_amdgcn_wmma_f32_16x16x32_f16(false, a, false, b2,
                                                        (short)0, acc[2], false, false);
        acc[3] = __builtin_amdgcn_wmma_f32_16x16x32_f16(false, a, false, b3,
                                                        (short)0, acc[3], false, false);
        a = an; b0 = c0; b1 = c1; b2 = c2; b3 = c3;
    }

    const int half = lane >> 4;
    const int col  = lane & 15;
#pragma unroll
    for (int t = 0; t < 4; ++t) {
        const int n = n0 + t * 16 + col;
        const float bn = bias ? bias[n] : 0.0f;
#pragma unroll
        for (int r = 0; r < 8; ++r) {
            const int m = m0 + r + 8 * half;
            const float v = acc[t][r] + bn;
            if (mode == 0) {          // [h][s][dk]
                const int h = n >> 6, dk = n & 63;
                out16[((size_t)h * S_ + m) * DK_ + dk] = (_Float16)v;
            } else if (mode == 1) {   // [h][dk][s]
                const int h = n >> 6, dk = n & 63;
                out16[((size_t)h * DK_ + dk) * S_ + m] = (_Float16)v;
            } else {                  // [m][n] f32
                out32[(size_t)m * N + n] = v;
            }
        }
    }
}

// ---- fused attention: one block per (head, 16 query rows) ------------------
// Block = 256 threads = 8 waves. LDS score tile: 16 x 2048 f32 (padded).
__global__ void __launch_bounds__(256, 1)
fused_attn(const _Float16* __restrict__ qh,   // [h][s][dk]
           const _Float16* __restrict__ kh,   // [h][s][dk]  (== Bt for QK^T)
           const _Float16* __restrict__ vT,   // [h][dk][s]  (== Bt for AV)
           const int* __restrict__ mask,
           float* __restrict__ attn,          // (H,S,S) output weights
           _Float16* __restrict__ ctx16) {    // [s][d]
    __shared__ float tile[16][TPAD];          // ~128.25 KB
    __shared__ float ctxLds[16][DK_];         // 4 KB

    const int h   = blockIdx.x >> 7;
    const int i0  = (blockIdx.x & 127) << 4;
    const int tid = threadIdx.x;
    const int wv  = tid >> 5;
    const int lane = tid & 31;
    const int row  = lane & 15;
    const int half = lane >> 4;

    const _Float16* Q  = qh + (size_t)h * S_ * DK_;
    const _Float16* Kt = kh + (size_t)h * S_ * DK_;
    const _Float16* Vt = vT + (size_t)h * DK_ * S_;

    // ---------- Phase 1: scores -> LDS (scale + mask applied) ----------
    v16h aq0 = load_a_f16(Q, i0,  0, DK_, lane);
    v16h aq1 = load_a_f16(Q, i0, 32, DK_, lane);
    for (int jt = wv; jt < S_ / 16; jt += 8) {         // 16 tiles per wave
        const int j0 = jt * 16;
        v16h b0 = load_bt_f16(Kt, j0,  0, DK_, lane);
        v16h b1 = load_bt_f16(Kt, j0, 32, DK_, lane);
        v8f acc = {};
        acc = __builtin_amdgcn_wmma_f32_16x16x32_f16(false, aq0, false, b0,
                                                     (short)0, acc, false, false);
        acc = __builtin_amdgcn_wmma_f32_16x16x32_f16(false, aq1, false, b1,
                                                     (short)0, acc, false, false);
        const int j = j0 + row;
        const bool msk = (mask[j] == 0);
#pragma unroll
        for (int r = 0; r < 8; ++r)
            tile[r + 8 * half][j] = msk ? -1e9f : acc[r] * 0.125f;
    }
    __syncthreads();

    // ---------- Phase 2: softmax rows in LDS; write weights out ----------
    // wave wv owns rows 2*wv, 2*wv+1; 32 lanes x 64 cols each
#pragma unroll
    for (int rr = 0; rr < 2; ++rr) {
        const int r = wv * 2 + rr;
        float mx = -3.4e38f;
        for (int t = 0; t < 64; ++t)
            mx = fmaxf(mx, tile[r][lane + t * 32]);
#pragma unroll
        for (int d = 16; d > 0; d >>= 1)
            mx = fmaxf(mx, __shfl_xor(mx, d, 32));
        float sum = 0.0f;
        for (int t = 0; t < 64; ++t) {
            const float e = expf(tile[r][lane + t * 32] - mx);
            tile[r][lane + t * 32] = e;
            sum += e;
        }
#pragma unroll
        for (int d = 16; d > 0; d >>= 1)
            sum += __shfl_xor(sum, d, 32);
        const float inv = 1.0f / sum;
        float* gout = attn + ((size_t)h * S_ + i0 + r) * S_;
        for (int t = 0; t < 64; ++t) {
            const float w = tile[r][lane + t * 32] * inv;
            tile[r][lane + t * 32] = w;
            gout[lane + t * 32] = w;
        }
    }

    // zero the ctx accumulator (independent of tile writes above)
#pragma unroll
    for (int z = 0; z < 4; ++z)
        ((float*)ctxLds)[tid + z * 256] = 0.0f;
    __syncthreads();

    // ---------- Phase 3: ctx = P @ V, K-range split across 8 waves ----------
    v8f acc[4] = {};
    const int jbeg = wv * (S_ / 8);
    for (int j0 = jbeg; j0 < jbeg + S_ / 8; j0 += 32) {
        // A fragment from LDS tile (f32 -> f16 in registers)
        v16h a;
        {
            const int cb = j0 + half * 8;
#pragma unroll
            for (int j = 0; j < 8; ++j) {
                a[j]     = (_Float16)tile[row][cb + j];
                a[j + 8] = (_Float16)tile[row][cb + j + 16];
            }
        }
#pragma unroll
        for (int t = 0; t < 4; ++t) {
            v16h b = load_bt_f16(Vt, t * 16, j0, S_, lane);
            acc[t] = __builtin_amdgcn_wmma_f32_16x16x32_f16(false, a, false, b,
                                                            (short)0, acc[t], false, false);
        }
    }
#pragma unroll
    for (int t = 0; t < 4; ++t)
#pragma unroll
        for (int r = 0; r < 8; ++r)
            atomicAdd(&ctxLds[r + 8 * half][t * 16 + row], acc[t][r]);
    __syncthreads();

    // ---------- write ctx tile ----------
#pragma unroll
    for (int z = 0; z < 4; ++z) {
        const int idx = tid + z * 256;
        const int m = idx >> 6, dk = idx & 63;
        ctx16[(size_t)(i0 + m) * D_ + h * DK_ + dk] = (_Float16)ctxLds[m][dk];
    }
}

// ---------------------------------------------------------------------------

extern "C" void kernel_launch(void* const* d_in, const int* in_sizes, int n_in,
                              void* d_out, int out_size, void* d_ws, size_t ws_size,
                              hipStream_t stream) {
    (void)in_sizes; (void)n_in; (void)out_size; (void)ws_size;

    const float* q    = (const float*)d_in[0];
    const float* k    = (const float*)d_in[1];
    const float* v    = (const float*)d_in[2];
    const int*   mask = (const int*)  d_in[3];
    const float* Wq   = (const float*)d_in[4];
    const float* bq   = (const float*)d_in[5];
    const float* Wk   = (const float*)d_in[6];
    const float* bk   = (const float*)d_in[7];
    const float* Wv   = (const float*)d_in[8];
    const float* bv   = (const float*)d_in[9];
    const float* Wo   = (const float*)d_in[10];
    const float* bo   = (const float*)d_in[11];

    float* out_p  = (float*)d_out;                       // (1,S,D)
    float* attn_p = (float*)d_out + (size_t)S_ * D_;     // (1,H,S,S)

    const size_t SD = (size_t)S_ * D_;   // 2M elems
    const size_t DD = (size_t)D_ * D_;   // 1M elems

    _Float16* w    = (_Float16*)d_ws;
    _Float16* q16  = w;            w += SD;
    _Float16* k16  = w;            w += SD;
    _Float16* v16  = w;            w += SD;
    _Float16* Wq16 = w;            w += DD;
    _Float16* Wk16 = w;            w += DD;
    _Float16* Wv16 = w;            w += DD;
    _Float16* Wo16 = w;            w += DD;
    _Float16* qh16 = w;            w += SD;   // [h][s][dk]
    _Float16* kh16 = w;            w += SD;   // [h][s][dk]
    _Float16* vT16 = w;            w += SD;   // [h][dk][s]
    _Float16* ctx16 = q16;                    // q16 dead after projections

    // 1) precision conversion
    cvt_f32_f16<<<4096, 256, 0, stream>>>(q,  q16,  (int)SD);
    cvt_f32_f16<<<4096, 256, 0, stream>>>(k,  k16,  (int)SD);
    cvt_f32_f16<<<4096, 256, 0, stream>>>(v,  v16,  (int)SD);
    cvt_f32_f16<<<2048, 256, 0, stream>>>(Wq, Wq16, (int)DD);
    cvt_f32_f16<<<2048, 256, 0, stream>>>(Wk, Wk16, (int)DD);
    cvt_f32_f16<<<2048, 256, 0, stream>>>(Wv, Wv16, (int)DD);
    cvt_f32_f16<<<2048, 256, 0, stream>>>(Wo, Wo16, (int)DD);

    // 2) projections: (S/16)*(D/64) = 2048 waves -> 256 blocks of 8 waves
    gemm_wmma<<<256, 256, 0, stream>>>(q16, Wq16, bq, qh16, nullptr,
                                       S_, D_, D_, 0);
    gemm_wmma<<<256, 256, 0, stream>>>(k16, Wk16, bk, kh16, nullptr,
                                       S_, D_, D_, 0);
    gemm_wmma<<<256, 256, 0, stream>>>(v16, Wv16, bv, vT16, nullptr,
                                       S_, D_, D_, 1);

    // 3) fused attention: one block per (head, 16-row tile)
    fused_attn<<<H_ * (S_ / 16), 256, 0, stream>>>(qh16, kh16, vT16, mask,
                                                   attn_p, ctx16);

    // 4) output projection (f32 + bias)
    gemm_wmma<<<256, 256, 0, stream>>>(ctx16, Wo16, bo, nullptr, out_p,
                                       S_, D_, D_, 2);
}
